// GRACE_39831526703449
// MI455X (gfx1250) — matrix operand order repaired and checked by reading, not compiled, not run on hardware
//
#include <hip/hip_runtime.h>
#include <math.h>

typedef __attribute__((ext_vector_type(16))) _Float16 v16h;
typedef __attribute__((ext_vector_type(8)))  _Float16 v8h;
typedef __attribute__((ext_vector_type(4)))  _Float16 v4h;
typedef __attribute__((ext_vector_type(8)))  float    v8f;

#define NUM_GRAPHS 64
#define GRAPH_SIZE 1024
#define PROJ_DIM   64
#define N_TOTAL    (NUM_GRAPHS * GRAPH_SIZE)   // 65536
#define INV_T      2.0f                        // 1 / 0.5

// LDS tile buffer geometry: 32 rows (16 z2-rows then 16 z1-rows), 128B of data
// per row padded to 144B (row-to-row bank rotation of 36 banks -> the 16 rows
// hit 16 distinct bank groups; conflict-free ds_load_b128).
#define ROW_BYTES   144
#define ROW_HALVES  72
#define BUF_BYTES   (32 * ROW_BYTES)           // 4608

// ---------------------------------------------------------------------------
// CDNA5 async copy: global -> LDS, 16 bytes per lane, tracked by ASYNCcnt.
// (CDNA5 ISA; VDST = LDS byte offset VGPR, GV mode)
// ---------------------------------------------------------------------------
__device__ __forceinline__ void async_cp16(unsigned int lds_off, const void* g)
{
    asm volatile("global_load_async_to_lds_b128 %0, %1, off"
                 :: "v"(lds_off), "v"((unsigned long long)(uintptr_t)g)
                 : "memory");
}
__device__ __forceinline__ void async_wait0()
{
    asm volatile("s_wait_asynccnt 0" ::: "memory");
}

// ---------------------------------------------------------------------------
// Kernel 0: zero the scalar output
// ---------------------------------------------------------------------------
__global__ void GRACE_zero(float* out) { out[0] = 0.0f; }

// ---------------------------------------------------------------------------
// Kernel 1: row-normalize z1/z2 (fp32 in), store f16 row-major (64/row).
// 16 lanes per row; shuffle-reduce sum of squares inside the 16-lane half.
// ---------------------------------------------------------------------------
__global__ __launch_bounds__(256) void GRACE_normalize(
    const float* __restrict__ z1, const float* __restrict__ z2,
    _Float16* __restrict__ h1, _Float16* __restrict__ h2)
{
    int gtid = blockIdx.x * 256 + threadIdx.x;
    int row  = gtid >> 4;
    int lane = gtid & 15;

    const float* src;
    _Float16*    dst;
    int r;
    if (row < N_TOTAL) { src = z1; dst = h1; r = row; }
    else               { src = z2; dst = h2; r = row - N_TOTAL; }

    const float4 v = ((const float4*)(src + (size_t)r * PROJ_DIM))[lane];
    float ss = v.x * v.x + v.y * v.y + v.z * v.z + v.w * v.w;
    ss += __shfl_xor(ss, 1, 32);
    ss += __shfl_xor(ss, 2, 32);
    ss += __shfl_xor(ss, 4, 32);
    ss += __shfl_xor(ss, 8, 32);
    float inv = rsqrtf(ss);

    v4h o;
    o[0] = (_Float16)(v.x * inv);
    o[1] = (_Float16)(v.y * inv);
    o[2] = (_Float16)(v.z * inv);
    o[3] = (_Float16)(v.w * inv);
    ((v4h*)(dst + (size_t)r * PROJ_DIM))[lane] = o;
}

// ---------------------------------------------------------------------------
// Kernel 2: per-row sum of exp(2*sim) over sim12 = z1 z2^T and sim11 = z1 z1^T
// within each graph.  Block = 128 threads (4 waves) owns a 128-row band of one
// graph; each wave owns 32 rows (two 16-row M-tiles).  The 16-column B panels
// (one tile of z2 + one tile of z1 per step) are async-copied into a
// double-buffered LDS stage once per block and consumed by all 4 waves:
//   - 8 v_wmma_f32_16x16x32_f16 per wave per step (K=64 -> 2 chained WMMAs,
//     2 sim matrices, 2 M-tiles)
//   - B operand layout (16-bit B 32x16): lane N = lane&15, half h holds
//     K = [8h..8h+7],[16+8h..23+8h] -> two 16B ds loads per operand.
//   - A layout (16-bit A 16x32): lane M, half h holds K=[8h..8h+7],[16+8h..].
//   - C layout: VGPR r, half h <-> row M = 8h + r, col N = lane&15.
// ---------------------------------------------------------------------------
__global__ __launch_bounds__(128) void GRACE_simsum(
    const _Float16* __restrict__ h1, const _Float16* __restrict__ h2,
    float* __restrict__ S)
{
    __shared__ _Float16 lds[2][32][ROW_HALVES];

    const int b    = blockIdx.x >> 3;              // graph id
    const int band = blockIdx.x & 7;               // 128-row band within graph
    const int wv   = threadIdx.x >> 5;             // wave in block: 0..3
    const int lane = threadIdx.x & 31;
    const int h    = lane >> 4;
    const int mrow = lane & 15;

    const int rows_base = b * GRAPH_SIZE + band * 128 + wv * 32;  // 2 M-tiles

    // ---- load the two A tiles (persist across all 64 column tiles) ----
    union A { v16h v; v8h p[2]; };
    A a0lo, a0hi, a1lo, a1hi;
    {
        const _Float16* ar0 = h1 + (size_t)(rows_base + mrow) * PROJ_DIM;
        const _Float16* ar1 = ar0 + 16 * PROJ_DIM;
        a0lo.p[0] = *(const v8h*)(ar0 +      8 * h);
        a0lo.p[1] = *(const v8h*)(ar0 + 16 + 8 * h);
        a0hi.p[0] = *(const v8h*)(ar0 + 32 + 8 * h);
        a0hi.p[1] = *(const v8h*)(ar0 + 48 + 8 * h);
        a1lo.p[0] = *(const v8h*)(ar1 +      8 * h);
        a1lo.p[1] = *(const v8h*)(ar1 + 16 + 8 * h);
        a1hi.p[0] = *(const v8h*)(ar1 + 32 + 8 * h);
        a1hi.p[1] = *(const v8h*)(ar1 + 48 + 8 * h);
    }

    // ---- per-thread async-copy slots: 2x16B per step ----
    const int t    = threadIdx.x;                  // 0..127
    const int crow = t >> 3;                       // tile row 0..15
    const int ccol = (t & 7) * 16;                 // byte column 0..112
    const unsigned int lbase = (unsigned int)(uintptr_t)&lds[0][0][0];
    const unsigned int l12   = lbase + crow * ROW_BYTES + ccol;
    const unsigned int l11   = l12 + 16 * ROW_BYTES;
    const char* g12 = (const char*)(h2 + (size_t)b * GRAPH_SIZE * PROJ_DIM)
                      + crow * 128 + ccol;
    const char* g11 = (const char*)(h1 + (size_t)b * GRAPH_SIZE * PROJ_DIM)
                      + crow * 128 + ccol;

    float s0[8], s1[8];
#pragma unroll
    for (int r = 0; r < 8; ++r) { s0[r] = 0.0f; s1[r] = 0.0f; }

    // prologue: fill buffer 0 with column tile 0
    async_cp16(l12, g12);
    async_cp16(l11, g11);

    for (int jt = 0; jt < 64; ++jt) {
        const int cur = jt & 1;
        async_wait0();            // this wave's async copies have landed
        __syncthreads();          // everyone's copies have landed

        if (jt + 1 < 64) {        // prefetch next tile into the other buffer
            const unsigned int bsel = (unsigned int)((jt + 1) & 1) * BUF_BYTES;
            const size_t goff = (size_t)(jt + 1) * 16 * 128;  // 16 rows * 128B
            async_cp16(l12 + bsel, g12 + goff);
            async_cp16(l11 + bsel, g11 + goff);
        }

        // ---- B operands from LDS (shared by both M-tiles) ----
        union A b12lo, b12hi, b11lo, b11hi;
        const _Float16* r12 = &lds[cur][mrow][0];
        const _Float16* r11 = &lds[cur][16 + mrow][0];
        b12lo.p[0] = *(const v8h*)(r12 + 8 * h);
        b12lo.p[1] = *(const v8h*)(r12 + 16 + 8 * h);
        b12hi.p[0] = *(const v8h*)(r12 + 32 + 8 * h);
        b12hi.p[1] = *(const v8h*)(r12 + 48 + 8 * h);
        b11lo.p[0] = *(const v8h*)(r11 + 8 * h);
        b11lo.p[1] = *(const v8h*)(r11 + 16 + 8 * h);
        b11hi.p[0] = *(const v8h*)(r11 + 32 + 8 * h);
        b11hi.p[1] = *(const v8h*)(r11 + 48 + 8 * h);

        // ---- M-tile 0 ----
        v8f c;
        c = (v8f){};
        c = __builtin_amdgcn_wmma_f32_16x16x32_f16(false, a0lo.v, false, b12lo.v,
                                                   (short)0, c, false, false);
        c = __builtin_amdgcn_wmma_f32_16x16x32_f16(false, a0hi.v, false, b12hi.v,
                                                   (short)0, c, false, false);
#pragma unroll
        for (int r = 0; r < 8; ++r) s0[r] += __expf(c[r] * INV_T);

        c = (v8f){};
        c = __builtin_amdgcn_wmma_f32_16x16x32_f16(false, a0lo.v, false, b11lo.v,
                                                   (short)0, c, false, false);
        c = __builtin_amdgcn_wmma_f32_16x16x32_f16(false, a0hi.v, false, b11hi.v,
                                                   (short)0, c, false, false);
#pragma unroll
        for (int r = 0; r < 8; ++r) s0[r] += __expf(c[r] * INV_T);

        // ---- M-tile 1 ----
        c = (v8f){};
        c = __builtin_amdgcn_wmma_f32_16x16x32_f16(false, a1lo.v, false, b12lo.v,
                                                   (short)0, c, false, false);
        c = __builtin_amdgcn_wmma_f32_16x16x32_f16(false, a1hi.v, false, b12hi.v,
                                                   (short)0, c, false, false);
#pragma unroll
        for (int r = 0; r < 8; ++r) s1[r] += __expf(c[r] * INV_T);

        c = (v8f){};
        c = __builtin_amdgcn_wmma_f32_16x16x32_f16(false, a1lo.v, false, b11lo.v,
                                                   (short)0, c, false, false);
        c = __builtin_amdgcn_wmma_f32_16x16x32_f16(false, a1hi.v, false, b11hi.v,
                                                   (short)0, c, false, false);
#pragma unroll
        for (int r = 0; r < 8; ++r) s1[r] += __expf(c[r] * INV_T);
    }

    // reduce the 16 partial column-sums within each half-wave
#pragma unroll
    for (int r = 0; r < 8; ++r) {
        s0[r] += __shfl_xor(s0[r], 1, 32);
        s0[r] += __shfl_xor(s0[r], 2, 32);
        s0[r] += __shfl_xor(s0[r], 4, 32);
        s0[r] += __shfl_xor(s0[r], 8, 32);
        s1[r] += __shfl_xor(s1[r], 1, 32);
        s1[r] += __shfl_xor(s1[r], 2, 32);
        s1[r] += __shfl_xor(s1[r], 4, 32);
        s1[r] += __shfl_xor(s1[r], 8, 32);
    }

    if (mrow == 0) {
#pragma unroll
        for (int r = 0; r < 8; ++r) {
            S[rows_base + 8 * h + r]      = s0[r];   // tile 0, row M = 8h + r
            S[rows_base + 16 + 8 * h + r] = s1[r];   // tile 1
        }
    }
}

// ---------------------------------------------------------------------------
// Kernel 3: loss_i = log(S[i] - e^2) - 2*<n1_i, n2_i>;  out += mean(loss)
// ---------------------------------------------------------------------------
__global__ __launch_bounds__(256) void GRACE_reduce(
    const _Float16* __restrict__ h1, const _Float16* __restrict__ h2,
    const float* __restrict__ S, float* __restrict__ out)
{
    __shared__ float sm[256];
    int i = blockIdx.x * 256 + threadIdx.x;

    const _Float16* p1 = h1 + (size_t)i * PROJ_DIM;
    const _Float16* p2 = h2 + (size_t)i * PROJ_DIM;
    float pos = 0.0f;
#pragma unroll
    for (int c = 0; c < 8; ++c) {
        v8h a = ((const v8h*)p1)[c];
        v8h bb = ((const v8h*)p2)[c];
#pragma unroll
        for (int k = 0; k < 8; ++k) pos += (float)a[k] * (float)bb[k];
    }
    pos *= INV_T;

    const float e2 = 7.389056099f;            // exp(2): removes diag of sim11
    float total = S[i] - e2;
    float loss  = __logf(total) - pos;

    sm[threadIdx.x] = loss;
    __syncthreads();
    for (int st = 128; st > 0; st >>= 1) {
        if (threadIdx.x < st) sm[threadIdx.x] += sm[threadIdx.x + st];
        __syncthreads();
    }
    if (threadIdx.x == 0)
        atomicAdd(out, sm[0] * (1.0f / (float)N_TOTAL));
}

// ---------------------------------------------------------------------------
extern "C" void kernel_launch(void* const* d_in, const int* in_sizes, int n_in,
                              void* d_out, int out_size, void* d_ws, size_t ws_size,
                              hipStream_t stream)
{
    (void)in_sizes; (void)n_in; (void)out_size; (void)ws_size;

    const float* z1 = (const float*)d_in[0];
    const float* z2 = (const float*)d_in[1];
    // d_in[2] = batch (int64), d_in[3] = num_graphs: fixed sorted layout, unused.

    float* out = (float*)d_out;

    char* ws = (char*)d_ws;
    _Float16* h1 = (_Float16*)ws;                                     // 8 MB
    _Float16* h2 = (_Float16*)(ws + (size_t)N_TOTAL * PROJ_DIM * 2);  // 8 MB
    float*    S  = (float*)(ws + (size_t)2 * N_TOTAL * PROJ_DIM * 2); // 256 KB

    GRACE_zero<<<1, 1, 0, stream>>>(out);

    GRACE_normalize<<<(2 * N_TOTAL * 16) / 256, 256, 0, stream>>>(z1, z2, h1, h2);

    // 64 graphs * 8 bands = 512 blocks, 4 waves each, 32 rows per wave
    GRACE_simsum<<<NUM_GRAPHS * 8, 128, 0, stream>>>(h1, h2, S);

    GRACE_reduce<<<N_TOTAL / 256, 256, 0, stream>>>(h1, h2, S, out);
}
